// DeepSeekSparseAttention_54288386621596
// MI455X (gfx1250) — matrix operand order repaired and checked by reading, not compiled, not run on hardware
//
#include <hip/hip_runtime.h>
#include <hip/hip_bf16.h>

// ---------------------------------------------------------------------------
// DeepSeek sparse attention for MI455X (gfx1250, wave32, WMMA).
// B=1, T=2048, D=1024, H=16, DH=64, DI=32, KSEL=32
// ---------------------------------------------------------------------------

#define T_SEQ 2048
#define D_MODEL 1024
#define N_HEADS 16
#define D_HEAD 64
#define D_IDX 32
#define K_SEL 32

typedef __attribute__((ext_vector_type(16))) __bf16          v16bf;
typedef __attribute__((ext_vector_type(16))) unsigned short  v16us;
typedef __attribute__((ext_vector_type(8)))  float           v8f;
typedef __attribute__((ext_vector_type(2)))  float           v2f;

union FragAB {
    uint4  q[2];   // two 16-byte chunks (16 x bf16 = 32 bytes)
    v16us  us;
    v16bf  bf;
};

__device__ __forceinline__ unsigned short f32_to_bf16_rne(float f) {
    unsigned int u = __float_as_uint(f);
    unsigned int r = u + 0x7fffu + ((u >> 16) & 1u);   // round-to-nearest-even
    return (unsigned short)(r >> 16);
}

// ---------------------------------------------------------------------------
// Y[M,N] = X[M,K] @ W[N,K]^T   (nn.Linear, bias=false), M=2048, N=K=1024.
// fp32 in / fp32 out, bf16 v_wmma_f32_16x16x32_bf16 inner product.
// 256 threads = 8 waves; block tile 128x128; wave tile 32x64 (2x4 WMMA tiles).
// ---------------------------------------------------------------------------
__global__ __launch_bounds__(256)
void gemm_xwt_bf16_wmma(const float* __restrict__ X,
                        const float* __restrict__ W,
                        float* __restrict__ Y) {
    __shared__ unsigned short As[128 * 32];   // [m][k] bf16
    __shared__ unsigned short Bs[128 * 32];   // [n][k] bf16

    const int bm   = blockIdx.x * 128;
    const int bn   = blockIdx.y * 128;
    const int tid  = threadIdx.x;
    const int wave = tid >> 5;
    const int lane = tid & 31;
    const int lhalf = lane >> 4;     // 0: lanes 0-15, 1: lanes 16-31
    const int lrow  = lane & 15;

    const int wm = (wave & 3) * 32;  // wave row offset  (2 x 16-row tiles)
    const int wn = (wave >> 2) * 64; // wave col offset  (4 x 16-col tiles)

    v8f acc[2][4] = {};

    for (int k0 = 0; k0 < D_MODEL; k0 += 32) {
        // ---- stage 128x32 fp32 -> bf16 into LDS for both X and W panels ----
#pragma unroll
        for (int i = 0; i < 4; ++i) {
            const int c  = tid + i * 256;      // 0..1023 float4 chunks
            const int r  = c >> 3;             // 0..127
            const int cc = (c & 7) * 4;        // 0..28

            float4 xv = *(const float4*)(X + (size_t)(bm + r) * D_MODEL + k0 + cc);
            unsigned int xp0 = (unsigned int)f32_to_bf16_rne(xv.x) |
                               ((unsigned int)f32_to_bf16_rne(xv.y) << 16);
            unsigned int xp1 = (unsigned int)f32_to_bf16_rne(xv.z) |
                               ((unsigned int)f32_to_bf16_rne(xv.w) << 16);
            *(uint2*)&As[r * 32 + cc] = make_uint2(xp0, xp1);

            float4 wv = *(const float4*)(W + (size_t)(bn + r) * D_MODEL + k0 + cc);
            unsigned int wp0 = (unsigned int)f32_to_bf16_rne(wv.x) |
                               ((unsigned int)f32_to_bf16_rne(wv.y) << 16);
            unsigned int wp1 = (unsigned int)f32_to_bf16_rne(wv.z) |
                               ((unsigned int)f32_to_bf16_rne(wv.w) << 16);
            *(uint2*)&Bs[r * 32 + cc] = make_uint2(wp0, wp1);
        }
        // hint the next X panel toward the caches (global_prefetch_b8)
        if (k0 + 32 < D_MODEL)
            __builtin_prefetch(X + (size_t)(bm + (tid >> 1)) * D_MODEL + k0 + 32, 0, 0);
        __syncthreads();

        // ---- load A fragments (16x32 bf16, ISA 7.12.2 wave32 layout) ----
        FragAB fa[2];
#pragma unroll
        for (int i = 0; i < 2; ++i) {
            const int arow = wm + i * 16 + lrow;
            // elems 0..7  = K 0..7  (lo half) / 8..15  (hi half)
            // elems 8..15 = K 16..23(lo half) / 24..31 (hi half)
            fa[i].q[0] = *(const uint4*)&As[arow * 32 + lhalf * 8];
            fa[i].q[1] = *(const uint4*)&As[arow * 32 + 16 + lhalf * 8];
        }
        // ---- load B fragments (32x16): lane holds K 0..7,16..23 / 8..15,24..31 of col n ----
        FragAB fb[4];
#pragma unroll
        for (int j = 0; j < 4; ++j) {
            const int nrow = wn + j * 16 + lrow;
            fb[j].q[0] = *(const uint4*)&Bs[nrow * 32 + lhalf * 8];
            fb[j].q[1] = *(const uint4*)&Bs[nrow * 32 + 16 + lhalf * 8];
        }

        // ---- 8 WMMAs per k-step per wave ----
#pragma unroll
        for (int i = 0; i < 2; ++i)
#pragma unroll
            for (int j = 0; j < 4; ++j)
                acc[i][j] = __builtin_amdgcn_wmma_f32_16x16x32_bf16(
                    false, fa[i].bf, false, fb[j].bf,
                    (short)0, acc[i][j], false, false);

        __syncthreads();
    }

    // ---- store: C/D layout: N = lane%16, M = 8*(lane/16) + vgpr ----
#pragma unroll
    for (int i = 0; i < 2; ++i) {
#pragma unroll
        for (int j = 0; j < 4; ++j) {
            const int col = bn + wn + j * 16 + lrow;
#pragma unroll
            for (int v = 0; v < 8; ++v) {
                const int row = bm + wm + i * 16 + lhalf * 8 + v;
                Y[(size_t)row * D_MODEL + col] = acc[i][j][v];
            }
        }
    }
}

// ---------------------------------------------------------------------------
// Lightning indexer projection: out[h,t,i] = sum_d QK[t, h*64+d] * Wp[h,d,i]
// ---------------------------------------------------------------------------
__global__ __launch_bounds__(256)
void indexer_proj(const float* __restrict__ QK,      // [T, D]
                  const float* __restrict__ Wp,      // [H, DH, DI]
                  float* __restrict__ out) {         // [H, T, DI]
    const int gid = blockIdx.x * 256 + threadIdx.x;
    if (gid >= N_HEADS * T_SEQ * D_IDX) return;
    const int h = gid / (T_SEQ * D_IDX);
    const int r = gid - h * (T_SEQ * D_IDX);
    const int t = r / D_IDX;
    const int i = r - t * D_IDX;

    const float* qrow = QK + (size_t)t * D_MODEL + h * D_HEAD;
    const float* wcol = Wp + (size_t)h * D_HEAD * D_IDX + i;
    float acc = 0.f;
#pragma unroll 8
    for (int d = 0; d < D_HEAD; ++d)
        acc += qrow[d] * wcol[(size_t)d * D_IDX];
    out[((size_t)h * T_SEQ + t) * D_IDX + i] = acc;
}

// ---------------------------------------------------------------------------
// Indexer score matrix via fp32 WMMA (16x16x4) + per-query top-32 selection.
// Block = (head h, 16-query tile). 8 waves:
//   phase 1: each wave computes 16 of the 128 key tiles -> LDS slab [16][2048]
//   phase 2: each wave owns 2 queries; 32 rounds of wave-local argmax
// Dynamic LDS: 16*2048*4 = 128 KB (fits in 320 KB/WGP).
// ---------------------------------------------------------------------------
__global__ __launch_bounds__(256)
void score_topk_kernel(const float* __restrict__ iq,     // [H, T, DI]
                       const float* __restrict__ ik,     // [H, T, DI]
                       int* __restrict__ indices) {      // [H, T, K_SEL]
    extern __shared__ float sc[];                        // [16][T_SEQ]

    const int qt   = blockIdx.x;          // query tile (0..127)
    const int h    = blockIdx.y;
    const int tid  = threadIdx.x;
    const int wave = tid >> 5;
    const int lane = tid & 31;
    const int lhalf = lane >> 4;
    const int lrow  = lane & 15;

    const float scale = 0.17677669529663689f;  // 1/sqrt(32)

    // A fragments: f32 16x4 layout -> lane holds M=lrow, K = step*4 + lhalf*2 + {0,1}
    const float* aq = iq + ((size_t)h * T_SEQ + (size_t)qt * 16 + lrow) * D_IDX;
    v2f afr[8];
#pragma unroll
    for (int s = 0; s < 8; ++s)
        afr[s] = *(const v2f*)(aq + s * 4 + lhalf * 2);

    // ---- phase 1: score tiles via v_wmma_f32_16x16x4_f32 ----
    for (int kt = wave; kt < T_SEQ / 16; kt += 8) {
        const float* bk = ik + ((size_t)h * T_SEQ + (size_t)kt * 16 + lrow) * D_IDX;
        v8f acc = {};
#pragma unroll
        for (int s = 0; s < 8; ++s) {
            v2f bfr = *(const v2f*)(bk + s * 4 + lhalf * 2);
            acc = __builtin_amdgcn_wmma_f32_16x16x4_f32(
                false, afr[s], false, bfr, (short)0, acc, false, false);
        }
        // C/D layout: N(key) = lrow, M(query) = lhalf*8 + v
#pragma unroll
        for (int v = 0; v < 8; ++v)
            sc[(lhalf * 8 + v) * T_SEQ + kt * 16 + lrow] = acc[v] * scale;
    }
    __syncthreads();

    // ---- phase 2: wave-local top-32 per query (2 queries per wave) ----
#pragma unroll
    for (int qi = 0; qi < 2; ++qi) {
        const int q = wave * 2 + qi;
        float* row = sc + q * T_SEQ;
        const int t = qt * 16 + q;
        int* outp = indices + ((size_t)h * T_SEQ + t) * K_SEL;

        for (int r = 0; r < K_SEL; ++r) {
            unsigned long long bk = 0ull;
            for (int s = lane; s < T_SEQ; s += 32) {
                unsigned int u = __float_as_uint(row[s]);
                u = (u & 0x80000000u) ? ~u : (u | 0x80000000u);  // monotonic key
                unsigned long long key =
                    ((unsigned long long)u << 32) | (unsigned int)(T_SEQ - 1 - s);
                if (key > bk) bk = key;
            }
#pragma unroll
            for (int off = 16; off > 0; off >>= 1) {
                unsigned long long o = __shfl_xor(bk, off, 32);
                if (o > bk) bk = o;
            }
            const int s = T_SEQ - 1 - (int)(bk & 0xFFFFFFFFu);
            if (lane == 0) outp[r] = s;
            if ((s & 31) == lane) row[s] = -__builtin_inff();  // owner lane clears
        }
    }
}

// ---------------------------------------------------------------------------
// Sparse attention over the 32 selected keys per (h,t). 64 threads = 2 waves.
// ---------------------------------------------------------------------------
__global__ __launch_bounds__(64)
void sparse_attn_kernel(const float* __restrict__ Q,       // [T, D]
                        const float* __restrict__ K,       // [T, D]
                        const float* __restrict__ V,       // [T, D]
                        const int*   __restrict__ indices, // [H, T, K_SEL]
                        float* __restrict__ attn_out) {    // [T, D]
    __shared__ float q[D_HEAD];
    __shared__ int   sel[K_SEL];
    __shared__ float w[K_SEL];

    const int t = blockIdx.x;
    const int h = blockIdx.y;
    const int tid = threadIdx.x;

    q[tid] = Q[(size_t)t * D_MODEL + h * D_HEAD + tid];
    if (tid < K_SEL) sel[tid] = indices[((size_t)h * T_SEQ + t) * K_SEL + tid];
    __syncthreads();

    if (tid < K_SEL) {
        const float* krow = K + (size_t)sel[tid] * D_MODEL + h * D_HEAD;
        float s = 0.f;
#pragma unroll 8
        for (int d = 0; d < D_HEAD; ++d) s += q[d] * krow[d];
        s *= 0.125f;  // 1/sqrt(64)

        float m = s;   // wave32 max-reduce
#pragma unroll
        for (int off = 16; off > 0; off >>= 1)
            m = fmaxf(m, __shfl_xor(m, off, 32));
        const float e = __expf(s - m);
        float sum = e;
#pragma unroll
        for (int off = 16; off > 0; off >>= 1)
            sum += __shfl_xor(sum, off, 32);
        w[tid] = e / sum;
    }
    __syncthreads();

    float acc = 0.f;
#pragma unroll
    for (int k = 0; k < K_SEL; ++k)
        acc += w[k] * V[(size_t)sel[k] * D_MODEL + h * D_HEAD + tid];
    attn_out[(size_t)t * D_MODEL + h * D_HEAD + tid] = acc;
}

// ---------------------------------------------------------------------------
// Launcher
// ---------------------------------------------------------------------------
extern "C" void kernel_launch(void* const* d_in, const int* in_sizes, int n_in,
                              void* d_out, int out_size, void* d_ws, size_t ws_size,
                              hipStream_t stream) {
    (void)in_sizes; (void)n_in; (void)out_size; (void)ws_size;

    const float* hidden = (const float*)d_in[0];   // [T, D]
    const float* Wq     = (const float*)d_in[1];   // [D, D]
    const float* Wk     = (const float*)d_in[2];
    const float* Wv     = (const float*)d_in[3];
    const float* Wo     = (const float*)d_in[4];
    const float* idx_wq = (const float*)d_in[5];   // [H, DH, DI]
    const float* idx_wk = (const float*)d_in[6];

    // workspace layout (fp32 unless noted)
    float* Qb   = (float*)d_ws;                          // T*D
    float* Kb   = Qb  + (size_t)T_SEQ * D_MODEL;         // T*D
    float* Vb   = Kb  + (size_t)T_SEQ * D_MODEL;         // T*D
    float* iqb  = Vb  + (size_t)T_SEQ * D_MODEL;         // H*T*DI
    float* ikb  = iqb + (size_t)N_HEADS * T_SEQ * D_IDX; // H*T*DI
    float* attn = ikb + (size_t)N_HEADS * T_SEQ * D_IDX; // T*D
    int*   idxb = (int*)(attn + (size_t)T_SEQ * D_MODEL);// H*T*K_SEL

    const dim3 gemm_grid(T_SEQ / 128, D_MODEL / 128);    // 16 x 8
    const dim3 gemm_blk(256);

    // 1) Q/K/V projections (bf16 WMMA)
    gemm_xwt_bf16_wmma<<<gemm_grid, gemm_blk, 0, stream>>>(hidden, Wq, Qb);
    gemm_xwt_bf16_wmma<<<gemm_grid, gemm_blk, 0, stream>>>(hidden, Wk, Kb);
    gemm_xwt_bf16_wmma<<<gemm_grid, gemm_blk, 0, stream>>>(hidden, Wv, Vb);

    // 2) Lightning indexer projections
    const int nproj = N_HEADS * T_SEQ * D_IDX;
    indexer_proj<<<(nproj + 255) / 256, 256, 0, stream>>>(Qb, idx_wq, iqb);
    indexer_proj<<<(nproj + 255) / 256, 256, 0, stream>>>(Kb, idx_wk, ikb);

    // 3) Score matrix (fp32 WMMA) + top-32 per (h, t)
    score_topk_kernel<<<dim3(T_SEQ / 16, N_HEADS), 256,
                        16 * T_SEQ * sizeof(float), stream>>>(iqb, ikb, idxb);

    // 4) Sparse attention over selected keys
    sparse_attn_kernel<<<dim3(T_SEQ, N_HEADS), 64, 0, stream>>>(Qb, Kb, Vb, idxb, attn);

    // 5) Output projection
    gemm_xwt_bf16_wmma<<<gemm_grid, gemm_blk, 0, stream>>>(attn, Wo, (float*)d_out);
}
